// MultiHeadAttention_82471962018565
// MI455X (gfx1250) — matrix operand order repaired and checked by reading, not compiled, not run on hardware
//
#include <hip/hip_runtime.h>

// ---------------------------------------------------------------------------
// MHA forward for gfx1250 (MI455X): bf16 WMMA GEMMs + flash attention, both
// with double-buffered async global->LDS staging (ASYNCcnt).
// B=2, T=2048, DM=2048, HQ=16, HKV=4, D=128, causal, RoPE base 10000.
// ---------------------------------------------------------------------------

#define BDIM 2
#define TDIM 2048
#define DMODEL 2048
#define HQN 16
#define HKVN 4
#define DHEAD 128
#define MROWS (BDIM * TDIM)          // 4096
#define MULT 0.08838834764831845f

typedef __attribute__((ext_vector_type(16))) __bf16       v16bf;
typedef __attribute__((ext_vector_type(8)))  float        v8f;
typedef __attribute__((ext_vector_type(4)))  unsigned int u32x4;

union BfOp { u32x4 q[2]; v16bf v; };

__device__ __forceinline__ unsigned short f2bf(float f) {
    unsigned int u = __float_as_uint(f);
    u += 0x7fffu + ((u >> 16) & 1u);          // round-to-nearest-even
    return (unsigned short)(u >> 16);
}
__device__ __forceinline__ float bf2f(unsigned short h) {
    return __uint_as_float(((unsigned int)h) << 16);
}

// Assemble a 16-element bf16 WMMA operand from two 16-byte chunks.
__device__ __forceinline__ v16bf load_op(const unsigned short* p0,
                                         const unsigned short* p1) {
    BfOp o;
    o.q[0] = *(const u32x4*)p0;
    o.q[1] = *(const u32x4*)p1;
    return o.v;
}

__device__ __forceinline__ v8f wmma_bf16(v16bf a, v16bf b, v8f c) {
    return __builtin_amdgcn_wmma_f32_16x16x32_bf16(
        false, a, false, b, (short)0, c, false, false);
}

__device__ __forceinline__ v8f zero8() {
    v8f z = {0.f, 0.f, 0.f, 0.f, 0.f, 0.f, 0.f, 0.f};
    return z;
}

// gfx1250 async copy: global -> LDS, 16 bytes per lane, tracked by ASYNCcnt.
__device__ __forceinline__ void async_load_b128(unsigned int lds_off,
                                                const unsigned short* gaddr) {
    asm volatile("global_load_async_to_lds_b128 %0, %1, off"
                 :: "v"(lds_off), "v"(gaddr) : "memory");
}
__device__ __forceinline__ void wait_async0() {
    asm volatile("s_wait_asynccnt 0" ::: "memory");
}

// --------------------------- conversion kernels ----------------------------

__global__ void f2bf_kernel(const float* __restrict__ in,
                            unsigned short* __restrict__ out, int n) {
    int i = blockIdx.x * blockDim.x + threadIdx.x;
    if (i < n) out[i] = f2bf(in[i]);
}

// in: [K, N] fp32 row-major  ->  out: [N, K] bf16 row-major
__global__ void transpose_bf_kernel(const float* __restrict__ in,
                                    unsigned short* __restrict__ out,
                                    int K, int N) {
    int i = blockIdx.x * blockDim.x + threadIdx.x;
    if (i >= N * K) return;
    int n = i / K, k = i - n * K;
    out[i] = f2bf(in[(size_t)k * N + n]);
}

// cos/sin tables: [T][64] fp32
__global__ void rope_table_kernel(float* __restrict__ cosT,
                                  float* __restrict__ sinT) {
    int t = blockIdx.x, j = threadIdx.x;      // 64 threads
    float inv = __powf(10000.0f, -(float)j * (1.0f / 64.0f));
    float ang = (float)t * inv;
    cosT[t * 64 + j] = cosf(ang);
    sinT[t * 64 + j] = sinf(ang);
}

// In-place RoPE on bf16 [B*H, T, D]; one thread per (bh, t, j<64) pair.
__global__ void rope_apply_kernel(unsigned short* __restrict__ x,
                                  const float* __restrict__ cosT,
                                  const float* __restrict__ sinT, int n) {
    int i = blockIdx.x * blockDim.x + threadIdx.x;
    if (i >= n) return;
    int j  = i & 63;
    int t  = (i >> 6) & (TDIM - 1);
    int bh = i >> 17;                          // 64 * 2048 = 2^17
    size_t base = ((size_t)bh * TDIM + t) * DHEAD;
    float c = cosT[t * 64 + j], s = sinT[t * 64 + j];
    float x1 = bf2f(x[base + j]);
    float x2 = bf2f(x[base + j + 64]);
    x[base + j]      = f2bf(x1 * c - x2 * s);
    x[base + j + 64] = f2bf(x2 * c + x1 * s);
}

// ------------------------------ WMMA GEMM ----------------------------------
// C[M,N] = A[M,K]bf16 @ Bt[N,K]bf16.
// block = 128 thr (4 waves); block tile 128x64; wave tile 32x64.
// B tile (64 rows x 32 k halves = 4KB) double-buffered in LDS via
// global_load_async_to_lds_b128; A software-pipelined in registers.
// mode 0: Cf fp32 row-major [M,N]
// mode 1: Cb bf16 [B, H, T, D]   (T=2048, D=128, b=m/T, h=n/D)
// mode 2: Cb bf16 [B, H, D, T]   (transposed heads, for V)
__global__ __launch_bounds__(128)
void gemm_bf16_kernel(const unsigned short* __restrict__ A,
                      const unsigned short* __restrict__ Bt,
                      float* __restrict__ Cf, unsigned short* __restrict__ Cb,
                      int M, int N, int K, int mode, int H) {
    __shared__ __align__(16) unsigned short Bs[2][64 * 32];

    const int tid  = threadIdx.x;
    const int lane = tid & 31;
    const int wid  = tid >> 5;
    const int l16  = lane & 15;
    const int hi   = lane >> 4;
    const int m0   = blockIdx.y * 128 + wid * 32;
    const int n0   = blockIdx.x * 64;

    // This thread's 32-byte slice of the 4KB B tile (64 rows x 64B).
    const int brow_ld = tid >> 1;             // 0..63
    const int bcol_ld = (tid & 1) * 16;       // halves
    const unsigned short* gB = Bt + (size_t)(n0 + brow_ld) * K + bcol_ld;
    // Low 32 bits of a flat shared pointer == LDS byte offset.
    const unsigned int ldsB0 =
        (unsigned int)(size_t)(&Bs[0][0]) + (unsigned int)tid * 32u;
    const unsigned int ldsB1 =
        (unsigned int)(size_t)(&Bs[1][0]) + (unsigned int)tid * 32u;

    v8f acc[2][4];
#pragma unroll
    for (int ms = 0; ms < 2; ++ms)
#pragma unroll
        for (int nt = 0; nt < 4; ++nt) acc[ms][nt] = zero8();

    const unsigned short* arow0 = A + (size_t)(m0 + l16) * K + hi * 8;
    const unsigned short* arow1 = arow0 + (size_t)16 * K;

    // Prologue: stage first B tile, preload first A operands.
    async_load_b128(ldsB0, gB);
    async_load_b128(ldsB0 + 16, gB + 8);
    v16bf a0 = load_op(arow0, arow0 + 16);
    v16bf a1 = load_op(arow1, arow1 + 16);

    int buf = 0;
    for (int kb = 0; kb < K; kb += 32, buf ^= 1) {
        wait_async0();            // B tile for this iteration resident
        __syncthreads();

        const int kn = (kb + 32 < K) ? kb + 32 : kb;
        if (kb + 32 < K) {        // stage next B tile into other buffer
            const unsigned int ldsN = buf ? ldsB0 : ldsB1;
            async_load_b128(ldsN, gB + kn);
            async_load_b128(ldsN + 16, gB + kn + 8);
        }
        // prefetch next A operands (overlaps with WMMA below)
        v16bf na0 = load_op(arow0 + kn, arow0 + kn + 16);
        v16bf na1 = load_op(arow1 + kn, arow1 + kn + 16);

#pragma unroll
        for (int nt = 0; nt < 4; ++nt) {
            const unsigned short* bp = &Bs[buf][(nt * 16 + l16) * 32 + hi * 8];
            v16bf b = load_op(bp, bp + 16);
            acc[0][nt] = wmma_bf16(a0, b, acc[0][nt]);
            acc[1][nt] = wmma_bf16(a1, b, acc[1][nt]);
        }
        __syncthreads();          // all reads of Bs[buf] done before overwrite
        a0 = na0;
        a1 = na1;
    }

#pragma unroll
    for (int ms = 0; ms < 2; ++ms) {
#pragma unroll
        for (int nt = 0; nt < 4; ++nt) {
#pragma unroll
            for (int r = 0; r < 8; ++r) {
                float v = acc[ms][nt][r];
                int mg = m0 + ms * 16 + r + hi * 8;
                int ng = n0 + nt * 16 + l16;
                if (mode == 0) {
                    Cf[(size_t)mg * N + ng] = v;
                } else {
                    int b = mg >> 11, t = mg & (TDIM - 1);
                    int h = ng >> 7,  d = ng & (DHEAD - 1);
                    if (mode == 1)
                        Cb[(((size_t)b * H + h) * TDIM + t) * DHEAD + d] = f2bf(v);
                    else
                        Cb[(((size_t)b * H + h) * DHEAD + d) * TDIM + t] = f2bf(v);
                }
            }
        }
    }
}

// --------------------------- flash attention -------------------------------
// grid = B*HQ*(T/64) blocks, 128 threads (4 waves); wave owns 16 query rows.
// K/V tiles for each 32-key step are shared by all 4 waves and staged into
// double-buffered LDS with async loads.  All waves iterate the block-uniform
// key range [0, qb64*64+64); past-diagonal steps are fully masked (P==0,
// alpha==1), so they are mathematical no-ops and barriers stay uniform.
// q: [B,HQ,T,D] bf16 (rope'd)   k: [B,HKV,T,D] bf16 (rope'd)
// vt: [B,HKV,D,T] bf16          o: [B,T,HQ*D] bf16
__global__ __launch_bounds__(128)
void attn_kernel(const unsigned short* __restrict__ q,
                 const unsigned short* __restrict__ k,
                 const unsigned short* __restrict__ vt,
                 unsigned short* __restrict__ o) {
    __shared__ __align__(16) unsigned short Ks[2][32 * 128];   // [key][d]
    __shared__ __align__(16) unsigned short Vs[2][128 * 32];   // [d][key]
    __shared__ __align__(16) unsigned short Plds[4][16 * 32];

    const int tid  = threadIdx.x;
    const int lane = tid & 31;
    const int wid  = tid >> 5;
    const int l16  = lane & 15;
    const int hi   = lane >> 4;

    const int blk   = blockIdx.x;
    const int qb64  = blk & 31;            // T/64 = 32
    const int h     = (blk >> 5) & 15;
    const int bi    = blk >> 9;
    const int hk    = h >> 2;              // GQA repeat_interleave: h//4
    const int qbase = qb64 * 64 + wid * 16;
    const int kend  = qb64 * 64 + 64;      // uniform across the block

    const unsigned short* kbasep = k + ((size_t)bi * HKVN + hk) * TDIM * DHEAD;
    const unsigned short* vbase  = vt + ((size_t)bi * HKVN + hk) * DHEAD * TDIM;

    // Async staging: 64B of K tile + 64B of V tile per thread per step.
    const int krow = tid >> 2;             // 0..31 (key within tile)
    const int kcol = (tid & 3) * 32;       // halves within D=128
    const unsigned short* gK = kbasep + (size_t)krow * DHEAD + kcol;
    const unsigned short* gV = vbase + (size_t)tid * TDIM;   // d = tid
    unsigned int ldsK[2], ldsV[2];
    ldsK[0] = (unsigned int)(size_t)(&Ks[0][0]) + (unsigned int)(krow * 256 + kcol * 2);
    ldsK[1] = (unsigned int)(size_t)(&Ks[1][0]) + (unsigned int)(krow * 256 + kcol * 2);
    ldsV[0] = (unsigned int)(size_t)(&Vs[0][0]) + (unsigned int)(tid * 64);
    ldsV[1] = (unsigned int)(size_t)(&Vs[1][0]) + (unsigned int)(tid * 64);

    auto stage = [&](int b, int kb0) {
        const unsigned short* gk = gK + (size_t)kb0 * DHEAD;
        async_load_b128(ldsK[b],      gk);
        async_load_b128(ldsK[b] + 16, gk + 8);
        async_load_b128(ldsK[b] + 32, gk + 16);
        async_load_b128(ldsK[b] + 48, gk + 24);
        const unsigned short* gv = gV + kb0;
        async_load_b128(ldsV[b],      gv);
        async_load_b128(ldsV[b] + 16, gv + 8);
        async_load_b128(ldsV[b] + 32, gv + 16);
        async_load_b128(ldsV[b] + 48, gv + 24);
    };

    // Q tile (16 x 128) as four A-operands, kept in registers.
    const unsigned short* qp =
        q + (((size_t)bi * HQN + h) * TDIM + (qbase + l16)) * DHEAD + hi * 8;
    v16bf qa[4];
#pragma unroll
    for (int ks = 0; ks < 4; ++ks)
        qa[ks] = load_op(qp + ks * 32, qp + ks * 32 + 16);

    v8f acc[8];
#pragma unroll
    for (int i = 0; i < 8; ++i) acc[i] = zero8();
    float mi[8], li[8];
#pragma unroll
    for (int r = 0; r < 8; ++r) { mi[r] = -1e30f; li[r] = 0.f; }

    unsigned short* pl = &Plds[wid][0];

    stage(0, 0);
    int buf = 0;
    for (int kb0 = 0; kb0 < kend; kb0 += 32, buf ^= 1) {
        wait_async0();              // this step's K/V tiles resident (own wave)
        __syncthreads();            // ... and everyone else's slices too
        if (kb0 + 32 < kend) stage(buf ^ 1, kb0 + 32);

        // ---- S = Q K^T (two 16-wide key tiles) from LDS ----
        v8f s0 = zero8(), s1 = zero8();
#pragma unroll
        for (int ks = 0; ks < 4; ++ks) {
            const unsigned short* kp0 = &Ks[buf][l16 * 128 + ks * 32 + hi * 8];
            const unsigned short* kp1 = &Ks[buf][(16 + l16) * 128 + ks * 32 + hi * 8];
            v16bf b0 = load_op(kp0, kp0 + 16);
            v16bf b1 = load_op(kp1, kp1 + 16);
            s0 = wmma_bf16(qa[ks], b0, s0);
            s1 = wmma_bf16(qa[ks], b1, s1);
        }

        // ---- causal mask + scale + online softmax ----
        const int qrow = qbase + hi * 8;
        const int key0 = kb0 + l16;
        float p0[8], p1[8], mn[8];
#pragma unroll
        for (int r = 0; r < 8; ++r) {
            int qr = qrow + r;
            float a0 = (key0      <= qr) ? s0[r] * MULT : -1e30f;
            float a1 = (key0 + 16 <= qr) ? s1[r] * MULT : -1e30f;
            p0[r] = a0; p1[r] = a1;
            mn[r] = fmaxf(a0, a1);
        }
#pragma unroll
        for (int r = 0; r < 8; ++r) {
#pragma unroll
            for (int off = 8; off >= 1; off >>= 1)
                mn[r] = fmaxf(mn[r], __shfl_xor(mn[r], off, 32));
            mn[r] = fmaxf(mn[r], mi[r]);
        }
        float alpha[8], rs[8];
#pragma unroll
        for (int r = 0; r < 8; ++r) {
            alpha[r] = __expf(mi[r] - mn[r]);
            p0[r] = __expf(p0[r] - mn[r]);
            p1[r] = __expf(p1[r] - mn[r]);
            rs[r] = p0[r] + p1[r];
            mi[r] = mn[r];
        }
#pragma unroll
        for (int r = 0; r < 8; ++r) {
#pragma unroll
            for (int off = 8; off >= 1; off >>= 1)
                rs[r] += __shfl_xor(rs[r], off, 32);
            li[r] = li[r] * alpha[r] + rs[r];
        }
#pragma unroll
        for (int dt = 0; dt < 8; ++dt)
#pragma unroll
            for (int r = 0; r < 8; ++r) acc[dt][r] *= alpha[r];

        // ---- P: C-layout -> A-layout via per-wave LDS tile (in-order) ----
        const int prow = hi * 8;
#pragma unroll
        for (int r = 0; r < 8; ++r) {
            pl[(prow + r) * 32 + l16]      = f2bf(p0[r]);
            pl[(prow + r) * 32 + 16 + l16] = f2bf(p1[r]);
        }
        const unsigned short* pr = pl + l16 * 32 + hi * 8;
        v16bf pa = load_op(pr, pr + 16);

        // ---- O += P @ V from LDS ----
#pragma unroll
        for (int dt = 0; dt < 8; ++dt) {
            const unsigned short* vp = &Vs[buf][(dt * 16 + l16) * 32 + hi * 8];
            v16bf bv = load_op(vp, vp + 16);
            acc[dt] = wmma_bf16(pa, bv, acc[dt]);
        }
        __syncthreads();            // reads of buf done before it is restaged
    }

    // ---- normalize + store to [B, T, HQ*D] bf16 ----
#pragma unroll
    for (int r = 0; r < 8; ++r) li[r] = 1.0f / li[r];
    const int tq0 = qbase + hi * 8;
#pragma unroll
    for (int dt = 0; dt < 8; ++dt) {
#pragma unroll
        for (int r = 0; r < 8; ++r) {
            size_t idx = ((size_t)bi * TDIM + tq0 + r) * (HQN * DHEAD) +
                         h * DHEAD + dt * 16 + l16;
            o[idx] = f2bf(acc[dt][r] * li[r]);
        }
    }
}

// ------------------------------- launcher ----------------------------------

extern "C" void kernel_launch(void* const* d_in, const int* in_sizes, int n_in,
                              void* d_out, int out_size, void* d_ws, size_t ws_size,
                              hipStream_t stream) {
    const float* x  = (const float*)d_in[0];
    // d_in[1] = causal mask (unused: mask applied analytically)
    const float* Wq = (const float*)d_in[2];
    const float* Wk = (const float*)d_in[3];
    const float* Wv = (const float*)d_in[4];
    const float* Wo = (const float*)d_in[5];
    float* out = (float*)d_out;

    char* p = (char*)d_ws;
    auto carve = [&](size_t bytes) -> char* {
        char* r = p;
        p += (bytes + 255) & ~(size_t)255;
        return r;
    };
    unsigned short* x_bf  = (unsigned short*)carve((size_t)MROWS * DMODEL * 2);
    unsigned short* wqt   = (unsigned short*)carve((size_t)DMODEL * DMODEL * 2);
    unsigned short* wkt   = (unsigned short*)carve((size_t)512 * DMODEL * 2);
    unsigned short* wvt   = (unsigned short*)carve((size_t)512 * DMODEL * 2);
    unsigned short* wot   = (unsigned short*)carve((size_t)DMODEL * DMODEL * 2);
    unsigned short* q_bf  = (unsigned short*)carve((size_t)BDIM * HQN * TDIM * DHEAD * 2);
    unsigned short* k_bf  = (unsigned short*)carve((size_t)BDIM * HKVN * TDIM * DHEAD * 2);
    unsigned short* vt_bf = (unsigned short*)carve((size_t)BDIM * HKVN * DHEAD * TDIM * 2);
    unsigned short* o_bf  = (unsigned short*)carve((size_t)BDIM * TDIM * HQN * DHEAD * 2);
    float* cosT = (float*)carve((size_t)TDIM * 64 * 4);
    float* sinT = (float*)carve((size_t)TDIM * 64 * 4);

    // 1) conversions / transposes
    {
        int n = MROWS * DMODEL;                      // 8,388,608
        f2bf_kernel<<<n / 256, 256, 0, stream>>>(x, x_bf, n);
    }
    transpose_bf_kernel<<<(DMODEL * DMODEL) / 256, 256, 0, stream>>>(Wq, wqt, DMODEL, DMODEL);
    transpose_bf_kernel<<<(512 * DMODEL) / 256, 256, 0, stream>>>(Wk, wkt, DMODEL, 512);
    transpose_bf_kernel<<<(512 * DMODEL) / 256, 256, 0, stream>>>(Wv, wvt, DMODEL, 512);
    transpose_bf_kernel<<<(DMODEL * DMODEL) / 256, 256, 0, stream>>>(Wo, wot, DMODEL, DMODEL);
    rope_table_kernel<<<TDIM, 64, 0, stream>>>(cosT, sinT);

    // 2) QKV projections (WMMA GEMMs, async-LDS staged B)
    gemm_bf16_kernel<<<dim3(DMODEL / 64, MROWS / 128), 128, 0, stream>>>(
        x_bf, wqt, nullptr, q_bf, MROWS, DMODEL, DMODEL, 1, HQN);
    gemm_bf16_kernel<<<dim3(512 / 64, MROWS / 128), 128, 0, stream>>>(
        x_bf, wkt, nullptr, k_bf, MROWS, 512, DMODEL, 1, HKVN);
    gemm_bf16_kernel<<<dim3(512 / 64, MROWS / 128), 128, 0, stream>>>(
        x_bf, wvt, nullptr, vt_bf, MROWS, 512, DMODEL, 2, HKVN);

    // 3) RoPE (in place)
    {
        int nq = BDIM * HQN * TDIM * 64;             // 4,194,304
        rope_apply_kernel<<<nq / 256, 256, 0, stream>>>(q_bf, cosT, sinT, nq);
        int nk = BDIM * HKVN * TDIM * 64;            // 1,048,576
        rope_apply_kernel<<<nk / 256, 256, 0, stream>>>(k_bf, cosT, sinT, nk);
    }

    // 4) causal flash attention (async-LDS staged K/V)
    attn_kernel<<<BDIM * HQN * (TDIM / 64), 128, 0, stream>>>(q_bf, k_bf, vt_bf, o_bf);

    // 5) output projection -> fp32
    gemm_bf16_kernel<<<dim3(DMODEL / 64, MROWS / 128), 128, 0, stream>>>(
        o_bf, wot, out, nullptr, MROWS, DMODEL, DMODEL, 0, 0);
}
